// SpatialTimeConv_30751965839469
// MI455X (gfx1250) — matrix operand order repaired and checked by reading, not compiled, not run on hardware
//
#include <hip/hip_runtime.h>
#include <hip/hip_bf16.h>

// MI455X / gfx1250, wave32. All heavy math routed through v_wmma_f32_16x16x32_f16.
// Block tile 64(M) x 128(N), K-step 32; 8 waves, each wave owns a 2x2 grid of
// 16x16 accumulators (4 WMMAs per K-step) for register-level A/B reuse.
// Staging: b128 global loads + b128/b16 LDS stores; B kept transposed in LDS so
// both A and B fragments are contiguous 16-byte LDS reads.

typedef __attribute__((ext_vector_type(16))) _Float16 v16h;
typedef __attribute__((ext_vector_type(8)))  _Float16 v8h;
typedef __attribute__((ext_vector_type(8)))  float    v8f;
typedef _Float16 f16_t;

#define PI_F 3.14159265358979323846f

__device__ __forceinline__ v16h cat16(v8h lo, v8h hi)
{
    return __builtin_shufflevector(lo, hi, 0, 1, 2, 3, 4, 5, 6, 7,
                                   8, 9, 10, 11, 12, 13, 14, 15);
}

// ---------------------------------------------------------------------------
// Generic batched WMMA GEMM:  C[b] = A[b] (MxK f16) @ B[b] (KxN f16) [+ bias]
// ---------------------------------------------------------------------------
__global__ __launch_bounds__(256)
void k_gemm_wmma(const f16_t* __restrict__ A, long long lda, long long sA,
                 const f16_t* __restrict__ B, long long ldb, long long sB,
                 float* __restrict__ C32, f16_t* __restrict__ C16,
                 long long ldc, long long sC,
                 const float* __restrict__ bias,
                 int M, int N, int K)
{
    __shared__ f16_t As[64][40];    // [m][k]  row = 80 B (16B-aligned slots)
    __shared__ f16_t Bt[128][40];   // [n][k]  transposed B tile

    const int tid  = threadIdx.x;
    const int lane = tid & 31;
    const int w    = tid >> 5;
    const int bz   = blockIdx.z;

    A += (size_t)bz * sA;
    B += (size_t)bz * sB;
    const size_t cbase = (size_t)bz * sC;

    const int wm   = w & 1;          // 32-row half of the block tile
    const int wn   = w >> 1;         // 32-col quarter (0..3)
    const int hs   = lane >> 4;      // half-wave select
    const int mrow = lane & 15;

    const int m0 = blockIdx.y * 64;
    const int n0 = blockIdx.x * 128;

    // per-thread staging coordinates
    const int ar = tid >> 2;             // A: row 0..63
    const int ac = (tid & 3) * 8;        // A: k-chunk 0/8/16/24
    const int br = tid >> 3;             // B: k-row 0..31
    const int bc = (tid & 7) * 16;       // B: n-chunk 0..112 (two v8h)

    const bool mn_full = (m0 + 64 <= M) && (n0 + 128 <= N);

    v8f acc00 = {}, acc01 = {}, acc10 = {}, acc11 = {};

    for (int k0 = 0; k0 < K; k0 += 32) {
        if (mn_full && (k0 + 32 <= K)) {
            // ---------------- fast path: all b128 ----------------
            const f16_t* ap = A + (size_t)(m0 + ar) * lda + (k0 + ac);
            const f16_t* bp = B + (size_t)(k0 + br) * ldb + (n0 + bc);
            if (k0 + 64 < K) {
                __builtin_prefetch(ap + 32, 0, 3);        // global_prefetch_b8
                __builtin_prefetch(bp + 32 * ldb, 0, 3);
            }
            v8h av  = *(const v8h*)ap;
            v8h bv0 = *(const v8h*)bp;
            v8h bv1 = *(const v8h*)(bp + 8);
            *(v8h*)&As[ar][ac] = av;
#pragma unroll
            for (int j = 0; j < 8; ++j) Bt[bc + j][br] = bv0[j];
#pragma unroll
            for (int j = 0; j < 8; ++j) Bt[bc + 8 + j][br] = bv1[j];
        } else {
            // ---------------- edge path ----------------
            v8h av = {};
            {
                int gm = m0 + ar;
                if (gm < M) {
                    const f16_t* ap = A + (size_t)gm * lda + (k0 + ac);
                    if (k0 + ac + 8 <= K) {
                        av = *(const v8h*)ap;
                    } else {
#pragma unroll
                        for (int j = 0; j < 8; ++j)
                            if (k0 + ac + j < K) av[j] = ap[j];
                    }
                }
                *(v8h*)&As[ar][ac] = av;
            }
            {
                int gk = k0 + br;
#pragma unroll
                for (int h = 0; h < 2; ++h) {
                    int cc = bc + h * 8;
                    v8h bv = {};
                    if (gk < K) {
                        const f16_t* bp = B + (size_t)gk * ldb + (n0 + cc);
                        if (n0 + cc + 8 <= N) {
                            bv = *(const v8h*)bp;
                        } else {
#pragma unroll
                            for (int j = 0; j < 8; ++j)
                                if (n0 + cc + j < N) bv[j] = bp[j];
                        }
                    }
#pragma unroll
                    for (int j = 0; j < 8; ++j) Bt[cc + j][br] = bv[j];
                }
            }
        }
        __syncthreads();

        // ---- fragments: contiguous b128 LDS reads ----
        const int ar0 = wm * 32 + mrow;
        v16h a0 = cat16(*(const v8h*)&As[ar0][8 * hs],
                        *(const v8h*)&As[ar0][16 + 8 * hs]);
        v16h a1 = cat16(*(const v8h*)&As[ar0 + 16][8 * hs],
                        *(const v8h*)&As[ar0 + 16][16 + 8 * hs]);

        const int bn0 = wn * 32 + mrow;
        v16h b0 = cat16(*(const v8h*)&Bt[bn0][8 * hs],
                        *(const v8h*)&Bt[bn0][16 + 8 * hs]);
        v16h b1 = cat16(*(const v8h*)&Bt[bn0 + 16][8 * hs],
                        *(const v8h*)&Bt[bn0 + 16][16 + 8 * hs]);

        acc00 = __builtin_amdgcn_wmma_f32_16x16x32_f16(false, a0, false, b0,
                                                       (short)0, acc00, false, false);
        acc01 = __builtin_amdgcn_wmma_f32_16x16x32_f16(false, a0, false, b1,
                                                       (short)0, acc01, false, false);
        acc10 = __builtin_amdgcn_wmma_f32_16x16x32_f16(false, a1, false, b0,
                                                       (short)0, acc10, false, false);
        acc11 = __builtin_amdgcn_wmma_f32_16x16x32_f16(false, a1, false, b1,
                                                       (short)0, acc11, false, false);
        __syncthreads();
    }

    // ---- write back (C layout: VGPR r -> M = r + 8*hs, lane&15 -> N) ----
    const int gn0 = n0 + wn * 32 + mrow;
    const int gn1 = gn0 + 16;
    const float bv0 = (bias && gn0 < N) ? bias[gn0] : 0.f;
    const float bv1 = (bias && gn1 < N) ? bias[gn1] : 0.f;
#pragma unroll
    for (int ai = 0; ai < 2; ++ai) {
        v8f accA = ai ? acc10 : acc00;
        v8f accB = ai ? acc11 : acc01;
#pragma unroll
        for (int r = 0; r < 8; ++r) {
            int gm = m0 + wm * 32 + ai * 16 + r + 8 * hs;
            if (gm >= M) continue;
            size_t ro = cbase + (size_t)gm * ldc;
            float v0 = accA[r] + bv0;
            float v1 = accB[r] + bv1;
            if (C32) {
                if (gn0 < N) C32[ro + gn0] = v0;
                if (gn1 < N) C32[ro + gn1] = v1;
            }
            if (C16) {
                if (gn0 < N) C16[ro + gn0] = (f16_t)v0;
                if (gn1 < N) C16[ro + gn1] = (f16_t)v1;
            }
        }
    }
}

// ---------------------------------------------------------------------------
// Small helper kernels
// ---------------------------------------------------------------------------
__global__ void k_f32_to_f16(const float* __restrict__ in, f16_t* __restrict__ out, int n)
{
    int i = blockIdx.x * blockDim.x + threadIdx.x;
    if (i < n) out[i] = (f16_t)in[i];
}

__global__ void k_zero_f16(f16_t* p, int n)
{
    int i = blockIdx.x * blockDim.x + threadIdx.x;
    if (i < n) p[i] = (f16_t)0.f;
}

// t = t_emb @ wd + bd  (512 outputs, K=512)
__global__ void k_tvec(const float* __restrict__ temb, const float* __restrict__ wd,
                       const float* __restrict__ bdv, float* __restrict__ tout)
{
    int j = blockIdx.x * blockDim.x + threadIdx.x;
    if (j >= 512) return;
    float s = bdv[j];
    for (int i = 0; i < 512; ++i) s += temb[i] * wd[i * 512 + j];
    tout[j] = s;
}

// wmix_h[i][j] = f16(wmix[i][j] * w_t[j]);  bsum[j] = b2[j] + b_t[j]
__global__ void k_prep(const float* __restrict__ wmix, const float* __restrict__ b2,
                       const float* __restrict__ tvec, f16_t* __restrict__ wmixh,
                       float* __restrict__ bsum)
{
    int i = blockIdx.x, j = threadIdx.x;
    wmixh[i * 256 + j] = (f16_t)(wmix[i * 256 + j] * tvec[j]);
    if (i == 0) bsum[j] = b2[j] + tvec[256 + j];
}

// MW quadrature weights for L=128 (length-255 extension folded to 128)
__global__ void k_qweights(float* __restrict__ q)
{
    __shared__ float wext[255];
    int k = threadIdx.x;
    if (k < 255) {
        float te = (2.f * k + 1.f) * PI_F / 255.f;
        float s = 0.f;
        for (int msi = -127; msi <= 127; ++msi) {
            float m = (float)msi;
            if (msi == 1 || msi == -1)   s += 0.5f * PI_F * sinf(te);
            else if ((msi & 1) == 0)     s += (2.f / (1.f - m * m)) * cosf(m * te);
        }
        wext[k] = s / 255.f;
    }
    __syncthreads();
    if (k < 128) {
        float v = wext[k];
        if (k < 127) v += wext[254 - k];
        q[k] = v;
    }
}

// DFT matrices (f16): Wre/Wim 127x256 (forward, only surviving m), Wi2 127x256
// (inverse real-part, K stacked [Re | -Im]).
__global__ void k_dftmats(f16_t* __restrict__ wre, f16_t* __restrict__ wim,
                          f16_t* __restrict__ wi2)
{
    int idx = blockIdx.x * blockDim.x + threadIdx.x;
    if (idx >= 127 * 256) return;
    int r = idx >> 8, c = idx & 255;
    float f  = (float)(r - 63);
    float th = -2.f * PI_F * f * (float)c / 255.f;
    wre[idx] = (f16_t)((c < 255) ? cosf(th) : 0.f);
    wim[idx] = (f16_t)((c < 255) ? sinf(th) : 0.f);
    float v = 0.f;
    if (c < 127)                  v =  cosf(2.f * PI_F * (float)(c - 63)       * (float)r / 127.f);
    else if (c >= 128 && c < 255) v = -sinf(2.f * PI_F * (float)(c - 128 - 63) * (float)r / 127.f);
    wi2[idx] = (f16_t)v;
}

// Spin-1 Wigner-d via 3-term recurrence; pointwise in (m, theta), each thread
// runs the l-recurrence privately. mode 0: scaled D_IN slice [mm=m-64][l][t]
// (Q, C_l, 2pi/255 folded in); mode 1: D_OUT^T [m][t][l] scaled by C_l.
__global__ void k_wigner(int L, int mode, const float* __restrict__ q,
                         f16_t* __restrict__ out)
{
    int idx = blockIdx.x * blockDim.x + threadIdx.x;
    int total = (2 * L - 1) * L;
    if (idx >= total) return;
    int m = idx / L;
    int t = idx - m * L;
    if (mode == 0 && (m < 64 || m >= 191)) return;

    float theta = (2.f * t + 1.f) * PI_F / (2.f * L - 1.f);
    float ct = cosf(theta), ch = cosf(0.5f * theta), sh = sinf(0.5f * theta);
    float msv = (float)(m - (L - 1));
    float scale_in = 2.f * PI_F / (2.f * L - 1.f);

    if (mode == 0) out[(size_t)(m - 64) * 8192 + t] = (f16_t)0.f;           // l = 0
    else           out[(size_t)m * 4096 + (size_t)t * 64] = (f16_t)0.f;

    float p2 = 0.f, p1 = 0.f;
    for (int l = 1; l < L; ++l) {
        float lf = (float)l;
        float num = (2.f * lf - 1.f) * ((lf - 1.f) * lf * ct + msv) * p1;
        if (l >= 2) {
            float a = (lf - 1.f) * (lf - 1.f);
            num -= lf * sqrtf(fmaxf((a - msv * msv) * (a - 1.f), 0.f)) * p2;
        }
        float den = (lf - 1.f) * sqrtf(fmaxf((lf * lf - msv * msv) * (lf * lf - 1.f), 0.f));
        float val = (den > 0.f) ? (num / den) : 0.f;
        if (l == 1 && m == L - 1) val = -sinf(theta) * 0.70710678118f;
        if (m == L - 1 + l || m == L - 1 - l) {
            float F = expf(0.5f * (lgammaf(2.f * lf + 1.f) - lgammaf(lf) - lgammaf(lf + 2.f)));
            if (m == L - 1 + l) {
                float sgn = (l & 1) ? 1.f : -1.f;                 // (-1)^(l+1) from (-sh)^(l+1)
                val = sgn * F * powf(ch, lf - 1.f) * powf(sh, lf + 1.f);
            } else {
                val = F * powf(ch, lf + 1.f) * powf(sh, lf - 1.f); // signs cancel
            }
        }
        if (mode == 0) {
            if (l < 64) {
                float cl = -sqrtf((2.f * lf + 1.f) / (4.f * PI_F));
                out[(size_t)(m - 64) * 8192 + (size_t)l * 128 + t] =
                    (f16_t)(val * q[t] * cl * scale_in);
            }
        } else {
            float cl = -sqrtf((2.f * lf + 1.f) / (4.f * PI_F));
            out[(size_t)m * 4096 + (size_t)t * 64 + l] = (f16_t)(val * cl);
        }
        p2 = p1; p1 = val;
    }
}

// ---------------------------------------------------------------------------
static inline void gemm(hipStream_t s,
                        const f16_t* A, long long lda, long long sA,
                        const f16_t* B, long long ldb, long long sB,
                        float* C32, f16_t* C16, long long ldc, long long sC,
                        const float* bias, int M, int N, int K, int batch)
{
    dim3 g((N + 127) / 128, (M + 63) / 64, batch);
    k_gemm_wmma<<<g, 256, 0, s>>>(A, lda, sA, B, ldb, sB, C32, C16, ldc, sC, bias, M, N, K);
}

// Workspace layout (bytes). Peak ~64 MB; intermediates fit L2 (192 MB).
static const size_t OFF_T     = 0;
static const size_t OFF_Q     = 4096;
static const size_t OFF_BSUM  = 8192;
static const size_t OFF_W1H   = 16384;                 // 64 KB
static const size_t OFF_W2H   = 81920;                 // 128 KB
static const size_t OFF_WMIXH = 212992;                // 128 KB
static const size_t OFF_WRE   = 344064;                // 64 KB slot
static const size_t OFF_WIM   = 409600;
static const size_t OFF_WI2   = 475136;
static const size_t OFF_DPIN  = 540672;                // 127*64*128 f16
static const size_t OFF_DOUTT = 2637824;               // 127*64*64 f16
static const size_t OFF_XH    = 4194304;               // 32640*128 f16
static const size_t OFF_FLMRE = 4194304;               // reuse x_h (dead after GEMM1)
static const size_t OFF_FLMIM = 8388608;
static const size_t OFF_H1    = 13631488;              // 32640*256 f16
static const size_t OFF_FTM2S = 13631488;              // reuse h1 (dead after GEMM2)
static const size_t OFF_G     = 23068672;              // 8128*256 f16
static const size_t OFF_H2    = 32505856;              // 32640*256 f16
static const size_t OFF_FTMRE = 50331648;              // 128*127*256 f16
static const size_t OFF_FTMIM = 58720256;

extern "C" void kernel_launch(void* const* d_in, const int* in_sizes, int n_in,
                              void* d_out, int out_size, void* d_ws, size_t ws_size,
                              hipStream_t stream)
{
    const float* x    = (const float*)d_in[0];   // (128,255,128)
    const float* temb = (const float*)d_in[1];   // (512,)
    const float* w1   = (const float*)d_in[2];   // (128,256)
    const float* b1   = (const float*)d_in[3];   // (256,)
    const float* wd   = (const float*)d_in[4];   // (512,512)
    const float* bdv  = (const float*)d_in[5];   // (512,)
    const float* wmix = (const float*)d_in[6];   // (256,256)
    const float* w2   = (const float*)d_in[7];   // (256,256)
    const float* b2   = (const float*)d_in[8];   // (256,)
    float* out = (float*)d_out;                  // (64,127,256) = 8128x256
    char*  ws  = (char*)d_ws;

    float* t_v   = (float*)(ws + OFF_T);
    float* q_v   = (float*)(ws + OFF_Q);
    float* bsum  = (float*)(ws + OFF_BSUM);
    f16_t* w1h   = (f16_t*)(ws + OFF_W1H);
    f16_t* w2h   = (f16_t*)(ws + OFF_W2H);
    f16_t* wmixh = (f16_t*)(ws + OFF_WMIXH);
    f16_t* wre   = (f16_t*)(ws + OFF_WRE);
    f16_t* wim   = (f16_t*)(ws + OFF_WIM);
    f16_t* wi2   = (f16_t*)(ws + OFF_WI2);
    f16_t* dpin  = (f16_t*)(ws + OFF_DPIN);
    f16_t* doutt = (f16_t*)(ws + OFF_DOUTT);
    f16_t* xh    = (f16_t*)(ws + OFF_XH);
    f16_t* flmre = (f16_t*)(ws + OFF_FLMRE);
    f16_t* flmim = (f16_t*)(ws + OFF_FLMIM);
    f16_t* h1    = (f16_t*)(ws + OFF_H1);
    f16_t* ftm2s = (f16_t*)(ws + OFF_FTM2S);
    f16_t* g     = (f16_t*)(ws + OFF_G);
    f16_t* h2    = (f16_t*)(ws + OFF_H2);
    f16_t* ftmre = (f16_t*)(ws + OFF_FTMRE);
    f16_t* ftmim = (f16_t*)(ws + OFF_FTMIM);

    // ---- precompute constants on device ----
    k_tvec<<<2, 256, 0, stream>>>(temb, wd, bdv, t_v);
    k_prep<<<256, 256, 0, stream>>>(wmix, b2, t_v, wmixh, bsum);
    k_qweights<<<1, 256, 0, stream>>>(q_v);
    k_f32_to_f16<<<(32640 * 128 + 255) / 256, 256, 0, stream>>>(x,  xh,  32640 * 128);
    k_f32_to_f16<<<(128 * 256 + 255) / 256, 256, 0, stream>>>(w1, w1h, 128 * 256);
    k_f32_to_f16<<<(256 * 256 + 255) / 256, 256, 0, stream>>>(w2, w2h, 256 * 256);
    k_dftmats<<<(127 * 256 + 255) / 256, 256, 0, stream>>>(wre, wim, wi2);
    k_wigner<<<(255 * 128 + 255) / 256, 256, 0, stream>>>(128, 0, q_v, dpin);
    k_wigner<<<(127 * 64 + 255) / 256, 256, 0, stream>>>(64, 1, q_v, doutt);

    // 1) h1 = x @ w1 + b1                           (32640 x 256, K=128)
    gemm(stream, xh, 128, 0, w1h, 256, 0, nullptr, h1, 256, 0, b1, 32640, 256, 128, 1);
    // 2) h2 = h1 @ (wmix * w_t)                     (32640 x 256, K=256)
    gemm(stream, h1, 256, 0, wmixh, 256, 0, nullptr, h2, 256, 0, nullptr, 32640, 256, 256, 1);
    // zero stacked [Re;Im] buffer (reuses h1 region, now dead; rows 127/255 stay 0)
    k_zero_f16<<<(64 * 256 * 256 + 255) / 256, 256, 0, stream>>>(ftm2s, 64 * 256 * 256);
    // 3) DFT (fftshifted, surviving m only), batched over theta t=0..127
    gemm(stream, wre, 256, 0, h2, 256, 255 * 256, nullptr, ftmre, 256, 127 * 256,
         nullptr, 127, 256, 255, 128);
    gemm(stream, wim, 256, 0, h2, 256, 255 * 256, nullptr, ftmim, 256, 127 * 256,
         nullptr, 127, 256, 255, 128);
    // 4) flm = D_in' @ ftm, batched over m (127)
    gemm(stream, dpin, 128, 64 * 128, ftmre, 127 * 256, 256, nullptr, flmre, 256, 64 * 256,
         nullptr, 64, 256, 128, 127);
    gemm(stream, dpin, 128, 64 * 128, ftmim, 127 * 256, 256, nullptr, flmim, 256, 64 * 256,
         nullptr, 64, 256, 128, 127);
    // 5) ftm2 = D_out'^T @ flm, batched over m; C written as [t][k][c] stacked Re/Im
    gemm(stream, doutt, 64, 64 * 64, flmre, 256, 64 * 256, nullptr, ftm2s, 256 * 256, 256,
         nullptr, 64, 256, 64, 127);
    gemm(stream, doutt, 64, 64 * 64, flmim, 256, 64 * 256, nullptr, ftm2s + 128 * 256,
         256 * 256, 256, nullptr, 64, 256, 64, 127);
    // 6) g = Re(iDFT): A = [Wre | -Wim] (127x256), batched over t (64)
    gemm(stream, wi2, 256, 0, ftm2s, 256, 256 * 256, nullptr, g, 256, 127 * 256,
         nullptr, 127, 256, 256, 64);
    // 7) out = g @ w2 + (b2 + b_t)                  (8128 x 256, K=256)
    gemm(stream, g, 256, 0, w2h, 256, 0, out, nullptr, 256, 0, bsum, 8128, 256, 256, 1);

    (void)in_sizes; (void)n_in; (void)out_size; (void)ws_size;
}
